// KAN_EBM_54073638256940
// MI455X (gfx1250) — matrix operand order repaired and checked by reading, not compiled, not run on hardware
//
#include <hip/hip_runtime.h>
#include <hip/hip_bf16.h>

typedef __attribute__((ext_vector_type(16))) _Float16 v16h;
typedef __attribute__((ext_vector_type(2)))  _Float16 h2;
typedef __attribute__((ext_vector_type(8)))  float    v8f;

#define TOKS_PER_BLK 128
#define LDSTRIDE 132          // 128 + 4 pad floats; 132 % 64 = 4 -> conflict-free strided row access
#define NBLK 512              // 65536 tokens / 128

// Expanded-K layout: each input feature owns 16 consecutive K slots
//   slot c=0 -> base value h ; c=1..12 -> hat basis j=c-1 ; c=13..15 -> zero-weight pad
#define L0_OFF      0         // din=4   Kpad=64   : 2*4*512   =   4096
#define L1_OFF   4096         // din=64  Kpad=1024 : 32*8*512  = 131072
#define L2_OFF 135168         // din=128 Kpad=2048 : 64*4*512  = 131072
#define L3_OFF 266240         // din=64  Kpad=1024 : 32*1*512  =  16384
#define WPACK_HALVES 282624
#define PARTIALS_OFF_BYTES 565248   // 282624 * 2 bytes, 256B aligned

__device__ __forceinline__ h2 u2h(unsigned u) { return __builtin_bit_cast(h2, u); }
__device__ __forceinline__ unsigned h2u(h2 h) { return __builtin_bit_cast(unsigned, h); }

// ---------------------------------------------------------------------------
// Pack bw (dout,din) + sw (dout,din,12) into f16 WMMA B-fragments.
// B layout: k = chunk*32 + 8*(lane>>4) + (e&7) + 16*(e>>3) ; n = nt*16 + (lane&15); i=k>>4, c=k&15.
// Pad slots (c>=13), pad rows (i>=din) and pad cols (n>=dout) get 0.0 weights,
// which is what lets the A-side skip zeroing its pad slots.
__global__ void kan_pack_weights(const float* __restrict__ bw,
                                 const float* __restrict__ sw,
                                 _Float16* __restrict__ dst,
                                 int din, int dout, int nnt, int nHalves) {
    int idx = blockIdx.x * 256 + threadIdx.x;
    if (idx >= nHalves) return;
    int e    = idx & 15;
    int lane = (idx >> 4) & 31;
    int f    = idx >> 9;
    int ch   = f / nnt;
    int nt   = f - ch * nnt;
    int k = ch * 32 + ((lane >> 4) << 3) + (e & 7) + ((e >> 3) << 4);
    int n = nt * 16 + (lane & 15);
    int i = k >> 4;
    int c = k & 15;
    float w = 0.f;
    if (n < dout && i < din && c < 13) {
        w = (c == 0) ? bw[n * din + i] : sw[(n * din + i) * 12 + (c - 1)];
    }
    dst[idx] = (_Float16)w;
}

// ---------------------------------------------------------------------------
// Scaled knots packed as f16 pairs: T_c = 2.5 * clamp(-1+(c-4)*0.4, -1, 1)
//   c:  0..4 -> -2.5 (0xC100) ; 5 -> -1.5 (0xBE00) ; 6 -> -0.5 (0xB800)
//       7 -> 0.5 (0x3800) ; 8 -> 1.5 (0x3E00) ; 9..15 -> 2.5 (0x4100)
// TLO[p] = {T(2p),T(2p+1)} (lanes 0-15, slots c=j), THI[p] = {T(8+2p),T(9+2p)}
__device__ __constant__ const unsigned kTLO[4] = {0xC100C100u, 0xC100C100u, 0xBE00C100u, 0x3800B800u};
__device__ __constant__ const unsigned kTHI[4] = {0x41003E00u, 0x41004100u, 0x41004100u, 0x41004100u};

// ---------------------------------------------------------------------------
// One KAN layer for one wave's 16-token tile; A synthesized with packed f16 math.
// A element e covers K = ch*32 + 8*hi + (e<8 ? e : e+8):
//   e=0..7  -> feature 2*ch,   slot c = 8*hi + e
//   e=8..15 -> feature 2*ch+1, slot c = 8*hi + (e-8)
// Pad slots (c>=13) are left holding their (finite) hat value: B holds 0 there.
template <int DIN, int NNT>
__device__ __forceinline__ void kan_layer_wave(const float* __restrict__ hin,
                                               float* __restrict__ hout,
                                               const _Float16* __restrict__ wpack,
                                               int lane) {
    constexpr int NCH = DIN / 2;     // Kpad = 16*DIN, 32 per chunk
    const int  m    = lane & 15;
    const bool ishi = (lane >> 4) != 0;
    const float* hrow = hin + m * LDSTRIDE;
    const v16h* wfrag = (const v16h*)wpack;

    v8f acc[NNT];
#pragma unroll
    for (int nt = 0; nt < NNT; ++nt) acc[nt] = v8f{0.f,0.f,0.f,0.f,0.f,0.f,0.f,0.f};

    const h2 one  = h2{(_Float16)1.f, (_Float16)1.f};
    const h2 zero = h2{(_Float16)0.f, (_Float16)0.f};

    for (int ch = 0; ch < NCH; ++ch) {
        float2 hp = *(const float2*)(hrow + 2 * ch);   // two features, one ds_load_b64
        union { v16h v; h2 h[8]; } A;
#pragma unroll
        for (int half = 0; half < 2; ++half) {
            float hv = half ? hp.y : hp.x;
            float xc = fminf(fmaxf(hv, -1.f), 1.f);
            _Float16 sh  = (_Float16)(2.5f * xc);
            _Float16 hvh = (_Float16)hv;               // base slot uses raw h
            h2 sv; sv[0] = sh; sv[1] = sh;
#pragma unroll
            for (int p = 0; p < 4; ++p) {
                h2 T  = u2h(ishi ? kTHI[p] : kTLO[p]); // one cndmask_b32
                h2 d  = sv - T;                        // v_pk_add_f16 (neg)
                h2 ad = u2h(h2u(d) & 0x7FFF7FFFu);     // packed abs
                h2 b  = one - ad;                      // v_pk_add_f16 (neg)
                h2 r  = __builtin_elementwise_max(b, zero); // v_pk_max_num_f16
                if (p == 0) r[0] = ishi ? r[0] : hvh;  // c==0 -> raw base value
                A.h[half * 4 + p] = r;
            }
        }
        v16h a = A.v;
#pragma unroll
        for (int nt = 0; nt < NNT; ++nt) {
            v16h b = wfrag[(ch * NNT + nt) * 32 + lane];
            acc[nt] = __builtin_amdgcn_wmma_f32_16x16x32_f16(
                false, a, false, b, (short)0, acc[nt], false, false);
        }
    }
    // C/D layout: VGPR v, lanes 0-15 -> row v, lanes 16-31 -> row v+8; col = nt*16 + (lane&15)
    const int hi = ishi ? 1 : 0;
#pragma unroll
    for (int nt = 0; nt < NNT; ++nt) {
#pragma unroll
        for (int v = 0; v < 8; ++v) {
            hout[(v + 8 * hi) * LDSTRIDE + nt * 16 + m] = acc[nt][v];
        }
    }
}

// ---------------------------------------------------------------------------
// LayerNorm with all 256 threads: 2 threads per row, E[x^2]-mu^2 variance.
__device__ __forceinline__ void layernorm_rows(float* buf, int dout,
                                               const float* __restrict__ gamma,
                                               const float* __restrict__ beta,
                                               int tid, float* lnS, float* lnQ) {
    const int row  = tid >> 1;
    const int half = tid & 1;
    const int hw   = dout >> 1;
    float* rowp = buf + row * LDSTRIDE + half * hw;
    const float* gp = gamma + half * hw;
    const float* bp = beta  + half * hw;

    float s = 0.f, q = 0.f;
    for (int n = 0; n < hw; ++n) { float v = rowp[n]; s += v; q += v * v; }
    lnS[tid] = s; lnQ[tid] = q;
    __syncthreads();
    float S = lnS[row * 2] + lnS[row * 2 + 1];
    float Q = lnQ[row * 2] + lnQ[row * 2 + 1];
    float mean = S / (float)dout;
    float var  = Q / (float)dout - mean * mean;
    float r = rsqrtf(var + 1e-5f);
    for (int n = 0; n < hw; ++n)
        rowp[n] = (rowp[n] - mean) * r * gp[n] + bp[n];
}

// ---------------------------------------------------------------------------
__global__ __launch_bounds__(256) void kan_fused(
    const float* __restrict__ u, const float* __restrict__ x,
    const _Float16* __restrict__ wpack,
    const float* __restrict__ g0, const float* __restrict__ be0,
    const float* __restrict__ g1, const float* __restrict__ be1,
    const float* __restrict__ g2, const float* __restrict__ be2,
    float* __restrict__ partials) {
    __shared__ float bufA[TOKS_PER_BLK * LDSTRIDE];
    __shared__ float bufB[TOKS_PER_BLK * LDSTRIDE];
    __shared__ float lnS[256];
    __shared__ float lnQ[256];

    const int tid  = threadIdx.x;
    const int wave = tid >> 5;
    const int lane = tid & 31;
    const int tok0 = blockIdx.x * TOKS_PER_BLK;
    const int woff = wave * 16 * LDSTRIDE;

    // Stage combined input [u | x] into bufA: (128 tokens x 4 feats)
    for (int idx = tid; idx < TOKS_PER_BLK * 4; idx += 256) {
        int t = idx >> 2, f = idx & 3;
        float v = (f == 0) ? u[tok0 + t] : x[(tok0 + t) * 3 + (f - 1)];
        bufA[t * LDSTRIDE + f] = v;
    }
    __syncthreads();

    // Layer 0: 4 -> 64
    kan_layer_wave<4, 4>(bufA + woff, bufB + woff, wpack + L0_OFF, lane);
    __syncthreads();
    layernorm_rows(bufB, 64, g0, be0, tid, lnS, lnQ);
    __syncthreads();

    // Layer 1: 64 -> 128
    kan_layer_wave<64, 8>(bufB + woff, bufA + woff, wpack + L1_OFF, lane);
    __syncthreads();
    layernorm_rows(bufA, 128, g1, be1, tid, lnS, lnQ);
    __syncthreads();

    // Layer 2: 128 -> 64
    kan_layer_wave<128, 4>(bufA + woff, bufB + woff, wpack + L2_OFF, lane);
    __syncthreads();
    layernorm_rows(bufB, 64, g2, be2, tid, lnS, lnQ);
    __syncthreads();

    // Layer 3: 64 -> 1 (N padded to 16; only col 0 carries signal)
    kan_layer_wave<64, 1>(bufB + woff, bufA + woff, wpack + L3_OFF, lane);
    __syncthreads();

    // Block-level sum of the 128 scalar outputs (reuse lnS as scratch)
    if (tid < TOKS_PER_BLK) lnS[tid] = bufA[tid * LDSTRIDE];
    __syncthreads();
    for (int s = TOKS_PER_BLK / 2; s > 0; s >>= 1) {
        if (tid < s) lnS[tid] += lnS[tid + s];
        __syncthreads();
    }
    if (tid == 0) partials[blockIdx.x] = lnS[0];
}

// ---------------------------------------------------------------------------
__global__ void kan_finalize(const float* __restrict__ partials,
                             float* __restrict__ out) {
    int b = threadIdx.x;
    if (b < 16) {
        float s = 0.f;
        for (int i = 0; i < 32; ++i) s += partials[b * 32 + i];
        out[b] = s / 4096.0f;
    }
}

// ---------------------------------------------------------------------------
extern "C" void kernel_launch(void* const* d_in, const int* in_sizes, int n_in,
                              void* d_out, int out_size, void* d_ws, size_t ws_size,
                              hipStream_t stream) {
    // setup_inputs() dict order:
    const float* u   = (const float*)d_in[0];
    const float* x   = (const float*)d_in[1];
    const float* bw0 = (const float*)d_in[2];
    const float* sw0 = (const float*)d_in[3];
    const float* g0  = (const float*)d_in[4];
    const float* be0 = (const float*)d_in[5];
    const float* bw1 = (const float*)d_in[6];
    const float* sw1 = (const float*)d_in[7];
    const float* g1  = (const float*)d_in[8];
    const float* be1 = (const float*)d_in[9];
    const float* bw2 = (const float*)d_in[10];
    const float* sw2 = (const float*)d_in[11];
    const float* g2  = (const float*)d_in[12];
    const float* be2 = (const float*)d_in[13];
    const float* bw3 = (const float*)d_in[14];
    const float* sw3 = (const float*)d_in[15];

    _Float16* wpack    = (_Float16*)d_ws;
    float*    partials = (float*)((char*)d_ws + PARTIALS_OFF_BYTES);

    // Pack weights (f32 -> f16, 16-slot expanded-K layout, B-fragment order)
    kan_pack_weights<<<(4096   + 255) / 256, 256, 0, stream>>>(bw0, sw0, wpack + L0_OFF,   4,  64, 4,   4096);
    kan_pack_weights<<<(131072 + 255) / 256, 256, 0, stream>>>(bw1, sw1, wpack + L1_OFF,  64, 128, 8, 131072);
    kan_pack_weights<<<(131072 + 255) / 256, 256, 0, stream>>>(bw2, sw2, wpack + L2_OFF, 128,  64, 4, 131072);
    kan_pack_weights<<<(16384  + 255) / 256, 256, 0, stream>>>(bw3, sw3, wpack + L3_OFF,  64,  16, 1,  16384);

    kan_fused<<<NBLK, 256, 0, stream>>>(u, x, wpack, g0, be0, g1, be1, g2, be2, partials);
    kan_finalize<<<1, 32, 0, stream>>>(partials, (float*)d_out);
}
